// HierarchicalTaskMoE_86165633893008
// MI455X (gfx1250) — matrix operand order repaired and checked by reading, not compiled, not run on hardware
//
#include <hip/hip_runtime.h>
#include <hip/hip_bf16.h>

#define HIDDEN 1024
#define INTER 2816
#define NUM_EXPERTS 8

typedef __attribute__((ext_vector_type(16))) __bf16 v16bf;
typedef __attribute__((ext_vector_type(8)))  float  v8f;
typedef __attribute__((ext_vector_type(4)))  int    v4i;

union AFrag { v16bf v; unsigned int u[8]; };
union BFrag { v16bf v; uint4 q[2]; };

constexpr int BM = 128;        // token tile
constexpr int BN = 64;         // output-feature tile
constexpr int KS = 64;         // K elements staged in LDS per barrier (2 WMMA steps)
constexpr int LDP = KS + 16;   // padded LDS row: 80 bf16 = 160B (16B aligned)

// ---- optional CDNA5 async global->LDS path (compile-safe fallback) ----------
#if __has_builtin(__builtin_amdgcn_global_load_async_to_lds_b128) && \
    __has_builtin(__builtin_amdgcn_s_wait_asynccnt)
#define HAVE_ASYNC_LDS 1
#else
#define HAVE_ASYNC_LDS 0
#endif

__device__ __forceinline__ void async_copy16(void* lds_dst, const void* gsrc) {
#if HAVE_ASYNC_LDS
  // AS3 inttoptr truncates generic addr to low 32 bits == LDS byte address.
  __builtin_amdgcn_global_load_async_to_lds_b128(
      (__attribute__((address_space(1))) v4i*)(uintptr_t)gsrc,
      (__attribute__((address_space(3))) v4i*)(uintptr_t)lds_dst,
      0, 0);
#else
  *(uint4*)lds_dst = *(const uint4*)gsrc;
#endif
}

__device__ __forceinline__ void async_wait() {
#if HAVE_ASYNC_LDS
  __builtin_amdgcn_s_wait_asynccnt(0);
#endif
}

// ---------------------------------------------------------------------------
// Kernel 1: task-level routing -> active_mask[8]
// ---------------------------------------------------------------------------
__global__ __launch_bounds__(256)
void task_route_kernel(const float* __restrict__ task_emb,
                       const int* __restrict__ task_id,
                       const float* __restrict__ router_w,
                       int* __restrict__ active_mask) {
  const int tid = threadIdx.x;
  const int e = tid >> 5, lane = tid & 31;
  const float* tv = task_emb + (size_t)task_id[0] * HIDDEN;
  float s = 0.f;
  for (int k = lane; k < HIDDEN; k += 32) s += router_w[(size_t)e * HIDDEN + k] * tv[k];
#pragma unroll
  for (int off = 16; off > 0; off >>= 1) s += __shfl_xor(s, off, 32);
  __shared__ float sc[NUM_EXPERTS];
  if (lane == 0) sc[e] = s;
  __syncthreads();
  if (tid == 0) {
    bool act[NUM_EXPERTS] = {}, sel[NUM_EXPERTS] = {};
    for (int t = 0; t < 4; ++t) {             // top-4 task experts (earliest on tie)
      int best = -1; float bv = 0.f;
      for (int i = 0; i < NUM_EXPERTS; ++i)
        if (!sel[i] && (best < 0 || sc[i] > bv)) { best = i; bv = sc[i]; }
      sel[best] = true; act[best] = true;
    }
    act[NUM_EXPERTS - 2] = true; act[NUM_EXPERTS - 1] = true;  // generalists
    for (int i = 0; i < NUM_EXPERTS; ++i) active_mask[i] = act[i] ? 1 : 0;
  }
}

// ---------------------------------------------------------------------------
// Kernel 2: token gating -> comb[N, 8] (one wave32 per token)
// ---------------------------------------------------------------------------
__global__ __launch_bounds__(256)
void gate_kernel(const float* __restrict__ X, const float* __restrict__ task_emb,
                 const int* __restrict__ task_id, const float* __restrict__ gw,
                 const int* __restrict__ active_mask, float* __restrict__ comb,
                 int Ntok) {
  const int lane = threadIdx.x & 31;
  const int token = blockIdx.x * 8 + (threadIdx.x >> 5);
  if (token >= Ntok) return;
  const float* tv = task_emb + (size_t)task_id[0] * HIDDEN;
  const float* x = X + (size_t)token * HIDDEN;
  float acc[NUM_EXPERTS];
#pragma unroll
  for (int e = 0; e < NUM_EXPERTS; ++e) acc[e] = 0.f;
  for (int k = lane; k < HIDDEN; k += 32) {
    float xv = x[k] + tv[k];
#pragma unroll
    for (int e = 0; e < NUM_EXPERTS; ++e) acc[e] += xv * gw[e * HIDDEN + k];
  }
#pragma unroll
  for (int e = 0; e < NUM_EXPERTS; ++e)
#pragma unroll
    for (int off = 16; off > 0; off >>= 1) acc[e] += __shfl_xor(acc[e], off, 32);
  if (lane == 0) {
    float logit[NUM_EXPERTS], m = -3.0e38f;
    for (int e = 0; e < NUM_EXPERTS; ++e) {
      logit[e] = active_mask[e] ? acc[e] : -1.0e9f;
      m = fmaxf(m, logit[e]);
    }
    float p[NUM_EXPERTS], denom = 0.f;
    for (int e = 0; e < NUM_EXPERTS; ++e) { p[e] = __expf(logit[e] - m); denom += p[e]; }
    for (int e = 0; e < NUM_EXPERTS; ++e) p[e] /= denom;
    int i1 = 0;
    for (int e = 1; e < NUM_EXPERTS; ++e) if (p[e] > p[i1]) i1 = e;
    int i2 = (i1 == 0) ? 1 : 0;
    for (int e = 0; e < NUM_EXPERTS; ++e) if (e != i1 && p[e] > p[i2]) i2 = e;
    float s = p[i1] + p[i2] + 1e-6f;
    float out[NUM_EXPERTS];
    for (int e = 0; e < NUM_EXPERTS; ++e) out[e] = 0.f;
    out[i1] = p[i1] / s; out[i2] = p[i2] / s;
    for (int e = 0; e < NUM_EXPERTS; ++e) comb[(size_t)token * NUM_EXPERTS + e] = out[e];
  }
}

// ---------------------------------------------------------------------------
// WMMA fragment loaders (per CDNA5 ISA 7.12.2 layouts, wave32)
// ---------------------------------------------------------------------------
__device__ __forceinline__ void load_a_frag(AFrag& a, const __bf16* row_ptr, int khalf) {
#pragma unroll
  for (int j = 0; j < 8; ++j) {
    const int kb = ((j & 4) ? 16 : 0) + khalf * 8 + (j & 3) * 2;  // ISA A 16x32 layout
    a.u[j] = *(const unsigned int*)(row_ptr + kb);
  }
}
__device__ __forceinline__ void load_b_frag(BFrag& b, const __bf16* col_ptr, int khalf) {
  const __bf16* p = col_ptr + khalf * 8;  // lanes 0-15: K 0..7,16..23; lanes 16-31: K 8..15,24..31
  b.q[0] = *(const uint4*)(p);
  b.q[1] = *(const uint4*)(p + 16);
}

// ---------------------------------------------------------------------------
// Kernel 3: per-expert fused gate/up GEMM + SwiGLU  ->  H[N, INTER] (bf16)
// ---------------------------------------------------------------------------
__global__ __launch_bounds__(256)
void moe_gemm1_swiglu(const float* __restrict__ X, const float* __restrict__ Gp,
                      const float* __restrict__ Up, __bf16* __restrict__ H) {
  __shared__ __align__(16) __bf16 As[BM][LDP];
  __shared__ __align__(16) __bf16 Gs[BN][LDP];
  __shared__ __align__(16) __bf16 Us[BN][LDP];

  const int tid = threadIdx.x;
  const int lane = tid & 31, wave = tid >> 5;
  const int wm = wave & 3, wn = wave >> 2;        // 4x2 wave grid, 32x32 per wave
  const int lrow = lane & 15, khalf = lane >> 4;

  const float* Abase = X  + (size_t)(blockIdx.y * BM) * HIDDEN;
  const float* Gbase = Gp + (size_t)(blockIdx.x * BN) * HIDDEN;
  const float* Ubase = Up + (size_t)(blockIdx.x * BN) * HIDDEN;

  v8f accG[2][2] = {};
  v8f accU[2][2] = {};

  for (int k0 = 0; k0 < HIDDEN; k0 += KS) {
    // tokens: 128x64 f32 -> bf16 LDS (8 float4 per thread)
#pragma unroll
    for (int i = 0; i < 8; ++i) {
      int idx = tid + i * 256, row = idx >> 4, c4 = (idx & 15) * 4;
      float4 v = *(const float4*)(Abase + (size_t)row * HIDDEN + k0 + c4);
      __bf16* d = &As[row][c4];
      d[0] = (__bf16)v.x; d[1] = (__bf16)v.y; d[2] = (__bf16)v.z; d[3] = (__bf16)v.w;
    }
    // weights: 2x 64x64 f32 -> bf16 LDS (4 float4 per thread each)
#pragma unroll
    for (int i = 0; i < 4; ++i) {
      int idx = tid + i * 256, row = idx >> 4, c4 = (idx & 15) * 4;
      float4 g = *(const float4*)(Gbase + (size_t)row * HIDDEN + k0 + c4);
      float4 u = *(const float4*)(Ubase + (size_t)row * HIDDEN + k0 + c4);
      __bf16* dg = &Gs[row][c4];
      dg[0] = (__bf16)g.x; dg[1] = (__bf16)g.y; dg[2] = (__bf16)g.z; dg[3] = (__bf16)g.w;
      __bf16* du = &Us[row][c4];
      du[0] = (__bf16)u.x; du[1] = (__bf16)u.y; du[2] = (__bf16)u.z; du[3] = (__bf16)u.w;
    }
    if (k0 + KS < HIDDEN) {  // global_prefetch_b8 on next weight K-tile
      __builtin_prefetch(Gbase + (size_t)(tid >> 2) * HIDDEN + k0 + KS, 0, 3);
      __builtin_prefetch(Ubase + (size_t)(tid >> 2) * HIDDEN + k0 + KS, 0, 3);
    }
    __syncthreads();

#pragma unroll
    for (int kk = 0; kk < KS; kk += 32) {
      AFrag a[2];
#pragma unroll
      for (int mi = 0; mi < 2; ++mi)
        load_a_frag(a[mi], &As[wm * 32 + mi * 16 + lrow][kk], khalf);
      BFrag gf[2], uf[2];
#pragma unroll
      for (int ni = 0; ni < 2; ++ni) {
        load_b_frag(gf[ni], &Gs[wn * 32 + ni * 16 + lrow][kk], khalf);
        load_b_frag(uf[ni], &Us[wn * 32 + ni * 16 + lrow][kk], khalf);
      }
#pragma unroll
      for (int mi = 0; mi < 2; ++mi)
#pragma unroll
        for (int ni = 0; ni < 2; ++ni) {
          accG[mi][ni] = __builtin_amdgcn_wmma_f32_16x16x32_bf16(
              false, a[mi].v, false, gf[ni].v, (short)0, accG[mi][ni], false, false);
          accU[mi][ni] = __builtin_amdgcn_wmma_f32_16x16x32_bf16(
              false, a[mi].v, false, uf[ni].v, (short)0, accU[mi][ni], false, false);
        }
    }
    __syncthreads();
  }

  // SwiGLU epilogue -> bf16 H. C/D frag: vgpr r -> row r + 8*(lane>=16), col = lane&15
  const int mo = khalf * 8;
#pragma unroll
  for (int mi = 0; mi < 2; ++mi)
#pragma unroll
    for (int ni = 0; ni < 2; ++ni)
#pragma unroll
      for (int r = 0; r < 8; ++r) {
        int row = blockIdx.y * BM + wm * 32 + mi * 16 + mo + r;
        int col = blockIdx.x * BN + wn * 32 + ni * 16 + lrow;
        float g = accG[mi][ni][r], u = accU[mi][ni][r];
        float h = (g / (1.0f + __expf(-g))) * u;
        H[(size_t)row * INTER + col] = (__bf16)h;
      }
}

// ---------------------------------------------------------------------------
// Kernel 4: per-expert down-projection, scaled by comb, accumulate into Y
// ---------------------------------------------------------------------------
__global__ __launch_bounds__(256)
void moe_gemm2_down(const __bf16* __restrict__ H, const float* __restrict__ Dp,
                    const float* __restrict__ comb, float* __restrict__ Y,
                    int e, int accumulate) {
  __shared__ __align__(16) __bf16 As[BM][LDP];
  __shared__ __align__(16) __bf16 Bs[BN][LDP];

  const int tid = threadIdx.x;
  const int lane = tid & 31, wave = tid >> 5;
  const int wm = wave & 3, wn = wave >> 2;
  const int lrow = lane & 15, khalf = lane >> 4;

  const __bf16* Hbase = H  + (size_t)(blockIdx.y * BM) * INTER;
  const float*  Dbase = Dp + (size_t)(blockIdx.x * BN) * INTER;

  v8f acc[2][2] = {};

  for (int k0 = 0; k0 < INTER; k0 += KS) {
    // H tile: 128x64 bf16, 4x 16B async copies per thread (ASYNCcnt path)
#pragma unroll
    for (int i = 0; i < 4; ++i) {
      int idx = tid + i * 256, row = idx >> 3, q = (idx & 7) * 8;
      async_copy16(&As[row][q], Hbase + (size_t)row * INTER + k0 + q);
    }
    // down_proj tile: 64x64 f32 -> bf16 (4 float4 per thread)
#pragma unroll
    for (int i = 0; i < 4; ++i) {
      int idx = tid + i * 256, row = idx >> 4, c4 = (idx & 15) * 4;
      float4 v = *(const float4*)(Dbase + (size_t)row * INTER + k0 + c4);
      __bf16* d = &Bs[row][c4];
      d[0] = (__bf16)v.x; d[1] = (__bf16)v.y; d[2] = (__bf16)v.z; d[3] = (__bf16)v.w;
    }
    if (k0 + KS < INTER)
      __builtin_prefetch(Dbase + (size_t)(tid >> 2) * INTER + k0 + KS, 0, 3);
    async_wait();
    __syncthreads();

#pragma unroll
    for (int kk = 0; kk < KS; kk += 32) {
      AFrag a[2];
#pragma unroll
      for (int mi = 0; mi < 2; ++mi)
        load_a_frag(a[mi], &As[wm * 32 + mi * 16 + lrow][kk], khalf);
      BFrag b[2];
#pragma unroll
      for (int ni = 0; ni < 2; ++ni)
        load_b_frag(b[ni], &Bs[wn * 32 + ni * 16 + lrow][kk], khalf);
#pragma unroll
      for (int mi = 0; mi < 2; ++mi)
#pragma unroll
        for (int ni = 0; ni < 2; ++ni)
          acc[mi][ni] = __builtin_amdgcn_wmma_f32_16x16x32_bf16(
              false, a[mi].v, false, b[ni].v, (short)0, acc[mi][ni], false, false);
    }
    __syncthreads();
  }

  const int mo = khalf * 8;
#pragma unroll
  for (int mi = 0; mi < 2; ++mi)
#pragma unroll
    for (int ni = 0; ni < 2; ++ni)
#pragma unroll
      for (int r = 0; r < 8; ++r) {
        int row = blockIdx.y * BM + wm * 32 + mi * 16 + mo + r;
        int col = blockIdx.x * BN + wn * 32 + ni * 16 + lrow;
        float scale = comb[(size_t)row * NUM_EXPERTS + e];
        float v = scale * acc[mi][ni][r];
        size_t o = (size_t)row * HIDDEN + col;
        if (accumulate) Y[o] += v; else Y[o] = v;
      }
}

// ---------------------------------------------------------------------------
extern "C" void kernel_launch(void* const* d_in, const int* in_sizes, int n_in,
                              void* d_out, int out_size, void* d_ws, size_t ws_size,
                              hipStream_t stream) {
  const float* X         = (const float*)d_in[0];
  const int*   task_id   = (const int*)d_in[1];
  const float* task_emb  = (const float*)d_in[2];
  const float* router_w  = (const float*)d_in[3];
  const float* gate_w    = (const float*)d_in[4];
  const float* gate_proj = (const float*)d_in[5];
  const float* up_proj   = (const float*)d_in[6];
  const float* down_proj = (const float*)d_in[7];
  float* Y = (float*)d_out;
  const int Ntok = in_sizes[0] / HIDDEN;  // 8192

  // workspace: [mask | comb f32 N*8 | H bf16 N*INTER]
  char* ws = (char*)d_ws;
  int*   active_mask = (int*)ws;
  float* comb = (float*)(ws + 256);
  size_t hoff = (256 + (size_t)Ntok * NUM_EXPERTS * sizeof(float) + 255) & ~(size_t)255;
  __bf16* H = (__bf16*)(ws + hoff);

  task_route_kernel<<<1, 256, 0, stream>>>(task_emb, task_id, router_w, active_mask);
  gate_kernel<<<(Ntok + 7) / 8, 256, 0, stream>>>(X, task_emb, task_id, gate_w,
                                                  active_mask, comb, Ntok);
  for (int e = 0; e < NUM_EXPERTS; ++e) {
    const float* Gp = gate_proj + (size_t)e * INTER * HIDDEN;
    const float* Up = up_proj   + (size_t)e * INTER * HIDDEN;
    const float* Dp = down_proj + (size_t)e * HIDDEN * INTER;
    moe_gemm1_swiglu<<<dim3(INTER / BN, Ntok / BM), 256, 0, stream>>>(X, Gp, Up, H);
    moe_gemm2_down<<<dim3(HIDDEN / BN, Ntok / BM), 256, 0, stream>>>(H, Dp, comb, Y,
                                                                     e, e > 0);
  }
}
